// graph_MHA_layer_64295660421248
// MI455X (gfx1250) — matrix operand order repaired and checked by reading, not compiled, not run on hardware
//
#include <hip/hip_runtime.h>
#include <hip/hip_bf16.h>
#include <cstdint>
#include <cstddef>

// ---------------------------------------------------------------------------
// Graph multi-head attention layer (GAT-style) for MI455X / gfx1250 (wave32).
//   Phase 1: fused QKV projection via V_WMMA_F32_16X16X32_F16 (f16 in, f32 acc)
//   Phase 2: per-edge scores -> segment softmax (atomic max/sum) -> weighted
//            V scatter (atomic f32 adds). Q/K/V (76.8 MB) are L2-resident.
// ---------------------------------------------------------------------------

typedef __attribute__((ext_vector_type(16))) _Float16 v16h;
typedef __attribute__((ext_vector_type(8)))  _Float16 v8h;
typedef __attribute__((ext_vector_type(8)))  float    v8f;

#define HIDDEN   128
#define HEADS    8
#define HEAD_DIM 16

// ------------------------------- utilities ---------------------------------

__global__ void k_cvt_f16(const float* __restrict__ in, _Float16* __restrict__ out, int n) {
  int i = blockIdx.x * blockDim.x + threadIdx.x;
  if (i < n) out[i] = (_Float16)in[i];
}

__global__ void k_fill_f32(float* __restrict__ p, float v, int n) {
  int i = blockIdx.x * blockDim.x + threadIdx.x;
  if (i < n) p[i] = v;
}

// Float atomic max via monotone integer orderings (works for mixed signs;
// buffer must be initialized to -inf).
__device__ __forceinline__ void atomicMaxF(float* addr, float val) {
  if (!(val < 0.0f)) {
    atomicMax((int*)addr, __float_as_int(val));
  } else {
    atomicMin((unsigned int*)addr, __float_as_uint(val));
  }
}

// --------------------------- phase 1: QKV GEMM -----------------------------
// One wave computes a 16-node x 128-out tile for one weight matrix.
// A: 16x32 f16 fragments of h-tile; B[k,n] = W[o=n][k] -> W rows contiguous.

__device__ __forceinline__ void qkv_project_one(
    const v16h a[4], const _Float16* __restrict__ W16,
    const float* __restrict__ bias, float* __restrict__ Out,
    int tile, int lane, int n_nodes)
{
  const int col    = lane & 15;
  const int ksel16 = (lane < 16) ? 0 : 8;    // B: lo half K 0..7/16..23, hi half 8..15/24..31
  const int mbase  = (lane < 16) ? 0 : 8;    // D: VGPR r -> row r (lo half) / r+8 (hi half)

  for (int ot = 0; ot < 8; ++ot) {
    v8f acc = {};
    const _Float16* wrow = W16 + (size_t)(ot * 16 + col) * HIDDEN;
    #pragma unroll
    for (int f = 0; f < 4; ++f) {
      v8h blo = *(const v8h*)(wrow + f * 32 + ksel16);       // 16B aligned
      v8h bhi = *(const v8h*)(wrow + f * 32 + 16 + ksel16);  // 16B aligned
      v16h b;
      #pragma unroll
      for (int e = 0; e < 8; ++e) { b[e] = blo[e]; b[8 + e] = bhi[e]; }
      acc = __builtin_amdgcn_wmma_f32_16x16x32_f16(
          /*neg_a=*/false, a[f], /*neg_b=*/false, b,
          /*c_mod=*/(short)0, acc, /*reuse_a=*/false, /*reuse_b=*/false);
    }
    const float bv = bias[ot * 16 + col];
    #pragma unroll
    for (int r = 0; r < 8; ++r) {
      int mrow = tile * 16 + mbase + r;
      if (mrow < n_nodes)
        Out[(size_t)mrow * HIDDEN + ot * 16 + col] = acc[r] + bv;
    }
  }
}

__global__ __launch_bounds__(256) void k_qkv_wmma(
    const _Float16* __restrict__ h16,
    const _Float16* __restrict__ w16,      // [3][128][128] f16, row-major (o,k)
    const float* __restrict__ bq, const float* __restrict__ bk,
    const float* __restrict__ bv,
    float* __restrict__ Q, float* __restrict__ K, float* __restrict__ V,
    int n_tiles, int n_nodes)
{
  const int lane = threadIdx.x & 31;
  const int tile = blockIdx.x * (blockDim.x >> 5) + (threadIdx.x >> 5);
  if (tile >= n_tiles) return;               // wave-uniform; EXEC stays all-1s for WMMA

  int row = tile * 16 + (lane & 15);
  if (row >= n_nodes) row = n_nodes - 1;
  const int ksel = (lane < 16) ? 0 : 8;      // A layout: lo half K 0..7/16..23, hi half 8..15/24..31
  const _Float16* hrow = h16 + (size_t)row * HIDDEN;

  v16h a[4];
  #pragma unroll
  for (int f = 0; f < 4; ++f) {
    v8h lo = *(const v8h*)(hrow + f * 32 + ksel);        // 16B aligned
    v8h hi = *(const v8h*)(hrow + f * 32 + 16 + ksel);   // 16B aligned
    #pragma unroll
    for (int e = 0; e < 8; ++e) { a[f][e] = lo[e]; a[f][8 + e] = hi[e]; }
  }

  qkv_project_one(a, w16,                     bq, Q, tile, lane, n_nodes);
  qkv_project_one(a, w16 + HIDDEN * HIDDEN,   bk, K, tile, lane, n_nodes);
  qkv_project_one(a, w16 + 2 * HIDDEN*HIDDEN, bv, V, tile, lane, n_nodes);
}

// --------------------------- phase 2: edge phase ---------------------------
// One thread per (edge, head).

__global__ void k_scores(const float* __restrict__ Q, const float* __restrict__ K,
                         const long long* __restrict__ src, const long long* __restrict__ dst,
                         float* __restrict__ scores, float* __restrict__ mbuf, int n_eh)
{
  int idx = blockIdx.x * blockDim.x + threadIdx.x;
  if (idx >= n_eh) return;
  const int e = idx >> 3, hh = idx & 7;
  const long long d = dst[e], s = src[e];
  const float4* q4 = (const float4*)(Q + (size_t)d * HIDDEN + hh * HEAD_DIM);
  const float4* k4 = (const float4*)(K + (size_t)s * HIDDEN + hh * HEAD_DIM);
  float acc = 0.0f;
  #pragma unroll
  for (int j = 0; j < 4; ++j) {
    float4 qv = q4[j], kv = k4[j];
    acc += qv.x * kv.x + qv.y * kv.y + qv.z * kv.z + qv.w * kv.w;
  }
  acc *= 0.25f;  // 1/sqrt(HEAD_DIM)
  scores[idx] = acc;
  atomicMaxF(mbuf + (size_t)d * HEADS + hh, acc);
}

__global__ void k_expsum(const long long* __restrict__ dst, const float* __restrict__ mbuf,
                         float* __restrict__ scores, float* __restrict__ denom, int n_eh)
{
  int idx = blockIdx.x * blockDim.x + threadIdx.x;
  if (idx >= n_eh) return;
  const int e = idx >> 3, hh = idx & 7;
  const long long d = dst[e];
  float ex = __expf(scores[idx] - mbuf[(size_t)d * HEADS + hh]);
  scores[idx] = ex;  // overwrite in place
  atomicAdd(denom + (size_t)d * HEADS + hh, ex);
}

__global__ void k_aggregate(const long long* __restrict__ src, const long long* __restrict__ dst,
                            const float* __restrict__ scores, const float* __restrict__ denom,
                            const float* __restrict__ V, float* __restrict__ out, int n_eh)
{
  int idx = blockIdx.x * blockDim.x + threadIdx.x;
  if (idx >= n_eh) return;
  const int e = idx >> 3, hh = idx & 7;
  const long long d = dst[e], s = src[e];
  const float attn = scores[idx] / denom[(size_t)d * HEADS + hh];
  const float4* v4 = (const float4*)(V + (size_t)s * HIDDEN + hh * HEAD_DIM);
  float* o = out + (size_t)d * HIDDEN + hh * HEAD_DIM;
  #pragma unroll
  for (int j = 0; j < 4; ++j) {
    float4 vv = v4[j];
    atomicAdd(o + 4 * j + 0, attn * vv.x);
    atomicAdd(o + 4 * j + 1, attn * vv.y);
    atomicAdd(o + 4 * j + 2, attn * vv.z);
    atomicAdd(o + 4 * j + 3, attn * vv.w);
  }
}

// ------------------------------- launcher ----------------------------------

extern "C" void kernel_launch(void* const* d_in, const int* in_sizes, int n_in,
                              void* d_out, int out_size, void* d_ws, size_t ws_size,
                              hipStream_t stream)
{
  const float*     h    = (const float*)d_in[0];
  const long long* src  = (const long long*)d_in[1];  // reference dtype int64
  const long long* dst  = (const long long*)d_in[2];
  const float*     WQ_w = (const float*)d_in[3];
  const float*     WQ_b = (const float*)d_in[4];
  const float*     WK_w = (const float*)d_in[5];
  const float*     WK_b = (const float*)d_in[6];
  const float*     WV_w = (const float*)d_in[7];
  const float*     WV_b = (const float*)d_in[8];

  const int n_nodes = in_sizes[0] / HIDDEN;
  const int n_edges = in_sizes[1];
  const int n_eh    = n_edges * HEADS;
  const int n_tiles = (n_nodes + 15) / 16;

  // ---- workspace carve (256B aligned) ----
  char*  base = (char*)d_ws;
  size_t off  = 0;
  auto carve = [&](size_t bytes) -> void* {
    off = (off + 255) & ~(size_t)255;
    void* p = base + off;
    off += bytes;
    return p;
  };
  _Float16* h16    = (_Float16*)carve((size_t)n_nodes * HIDDEN * sizeof(_Float16));
  _Float16* w16    = (_Float16*)carve((size_t)3 * HIDDEN * HIDDEN * sizeof(_Float16));
  float*    Q      = (float*)carve((size_t)n_nodes * HIDDEN * sizeof(float));
  float*    K      = (float*)carve((size_t)n_nodes * HIDDEN * sizeof(float));
  float*    V      = (float*)carve((size_t)n_nodes * HIDDEN * sizeof(float));
  float*    scores = (float*)carve((size_t)n_eh * sizeof(float));
  float*    mbuf   = (float*)carve((size_t)n_nodes * HEADS * sizeof(float));
  float*    denom  = (float*)carve((size_t)n_nodes * HEADS * sizeof(float));
  (void)ws_size; (void)n_in;

  const int T = 256;

  // convert h and weights to f16 for WMMA
  k_cvt_f16<<<(n_nodes * HIDDEN + T - 1) / T, T, 0, stream>>>(h, h16, n_nodes * HIDDEN);
  k_cvt_f16<<<(HIDDEN * HIDDEN + T - 1) / T, T, 0, stream>>>(WQ_w, w16, HIDDEN * HIDDEN);
  k_cvt_f16<<<(HIDDEN * HIDDEN + T - 1) / T, T, 0, stream>>>(WK_w, w16 + HIDDEN * HIDDEN, HIDDEN * HIDDEN);
  k_cvt_f16<<<(HIDDEN * HIDDEN + T - 1) / T, T, 0, stream>>>(WV_w, w16 + 2 * HIDDEN * HIDDEN, HIDDEN * HIDDEN);

  // fused QKV projection (8 waves / block, 1 wave = 16-node tile)
  k_qkv_wmma<<<(n_tiles + 7) / 8, 256, 0, stream>>>(
      h16, w16, WQ_b, WK_b, WV_b, Q, K, V, n_tiles, n_nodes);

  // init softmax accumulators + output (every launch: deterministic)
  k_fill_f32<<<(n_nodes * HEADS + T - 1) / T, T, 0, stream>>>(mbuf, -__builtin_inff(), n_nodes * HEADS);
  hipMemsetAsync(denom, 0, (size_t)n_nodes * HEADS * sizeof(float), stream);
  hipMemsetAsync(d_out, 0, (size_t)out_size * sizeof(float), stream);

  // edge phase
  k_scores<<<(n_eh + T - 1) / T, T, 0, stream>>>(Q, K, src, dst, scores, mbuf, n_eh);
  k_expsum<<<(n_eh + T - 1) / T, T, 0, stream>>>(dst, mbuf, scores, denom, n_eh);
  k_aggregate<<<(n_eh + T - 1) / T, T, 0, stream>>>(src, dst, scores, denom, V, (float*)d_out, n_eh);
}
